// AtomAttentionDecoder_2963527434500
// MI455X (gfx1250) — matrix operand order repaired and checked by reading, not compiled, not run on hardware
//
#include <hip/hip_runtime.h>
#include <hip/hip_bf16.h>
#include <cstdint>

typedef __attribute__((ext_vector_type(16))) __bf16 v16bf;
typedef __attribute__((ext_vector_type(8)))  float  v8f;
typedef __attribute__((ext_vector_type(4)))  unsigned int u32x4;
typedef __attribute__((ext_vector_type(8)))  int    i32x8;
typedef __attribute__((ext_vector_type(4)))  int    i32x4;

#define N_TOKEN 2048
#define N_ATOM  16384
#define C_TOKEN 768
#define C_ATOM  128
#define HH 4
#define DD 32
#define LL 3
#define NQ 32
#define NK 128
#define NB 512

__device__ __forceinline__ float sigf(float x) { return 1.0f / (1.0f + __expf(-x)); }

// ---------------------------------------------------------------------------
// TDM: DMA a 2D bf16 tile (tile_elems x rows, row stride = stride_elems) from
// global into LDS, inserting 4 DWORDs of LDS padding after every 64 DWORDs
// (= one 128-elem bf16 row) so the LDS image has a 136-half row pitch.
// D# layout per CDNA5 ISA ch.8 (group0 128b, group1 256b; groups 2/3 zero -> 2D).
// This toolchain's builtin takes 6 args (extra i32x8 group + cpol).
// ---------------------------------------------------------------------------
__device__ __forceinline__ void tdm_load_tile(void* ldsDst, const void* gsrc,
                                              int tile_elems, int rows, int stride_elems) {
  u32x4 g0; i32x8 g1; i32x4 gz; i32x8 gz8;
  gz[0] = 0; gz[1] = 0; gz[2] = 0; gz[3] = 0;
  gz8[0] = 0; gz8[1] = 0; gz8[2] = 0; gz8[3] = 0;
  gz8[4] = 0; gz8[5] = 0; gz8[6] = 0; gz8[7] = 0;
  unsigned long long ga = (unsigned long long)(uintptr_t)gsrc;
  g0[0] = 1u;                                                   // count=1 (valid D#)
  g0[1] = (unsigned)(unsigned long long)(uintptr_t)ldsDst;      // lds_addr
  g0[2] = (unsigned)ga;                                         // global_addr[31:0]
  g0[3] = (unsigned)((ga >> 32) & 0x01FFFFFFull) | (2u << 30);  // addr[56:32] | type=2
  g1[0] = (int)((1u << 16)      // data_size = 2 bytes
              | (1u << 20)      // pad_enable
              | (5u << 22)      // pad_interval: 64 DWORDs
              | (3u << 25));    // pad_amount: 4 DWORDs
  g1[1] = 0;                    // atomic_barrier=0; tensor_dim0[15:0]=0 (dim0 = 1<<20)
  g1[2] = 0x0010;               // tensor_dim0[31:16]=0x10; tensor_dim1[15:0]=0
  g1[3] = (int)(((unsigned)tile_elems << 16) | 0x0010u); // tensor_dim1 hi; tile_dim0
  g1[4] = rows;                 // tile_dim1 = rows; tile_dim2 = 0 (2D)
  g1[5] = stride_elems;         // tensor_dim0_stride[31:0]
  g1[6] = 0;
  g1[7] = 0;
  __builtin_amdgcn_tensor_load_to_lds(g0, g1, gz, gz, gz8, 0);
}

// ---------------------------------------------------------------------------
// Generic WMMA GEMM:  C[M,N] = epilogue( A[M,K](bf16) @ Bt[N,K](bf16)^T + bias )
// Bt is the pre-transposed, pre-scaled bf16 weight (prepared once by k_wprep).
// M%64==0, N%128==0, K%128==0 for all call sites.
// Tile 64x128 per 256-thread block (8 waves, 4 wmma tiles each); tiles staged
// by the Tensor Data Mover (wave0: B, wave1: A), compute is pure WMMA.
// EPI: 0=f32  1=bf16  2=sigmoid->f32  3=sig(aux1)*acc->f32
//      4=sig(aux1)*acc+aux2->f32      5=silu(aux1)*acc->bf16
// ---------------------------------------------------------------------------
template<int EPI>
__global__ __launch_bounds__(256) void gemm_bf16(
    const __bf16* __restrict__ A,  int lda,
    const __bf16* __restrict__ Bt, int ldb,
    const float*  __restrict__ bias,
    const float*  __restrict__ aux1,
    const float*  __restrict__ aux2,
    void*         __restrict__ Cout,
    int M, int N, int K)
{
  __shared__ __bf16 lA [64][136];    // [m][k]
  __shared__ __bf16 lBt[128][136];   // [n][k]

  const int tid   = threadIdx.x;
  const int wv    = tid >> 5;
  const int lane  = tid & 31;
  const int nlane = lane & 15;
  const int khalf = lane >> 4;
  const int rblk  = wv & 3;           // row block within tile (16 rows each)
  const int cbase = (wv >> 2) * 64;   // 0 or 64

  const int tileM = blockIdx.x * 64;
  const int tileN = blockIdx.y * 128;

  v8f acc[4] = {};

  for (int k0 = 0; k0 < K; k0 += 128) {
    __syncthreads();                  // previous tile fully consumed
    if (wv == 0) {
      tdm_load_tile(&lBt[0][0], Bt + (size_t)tileN * ldb + k0, 128, 128, ldb);
      __builtin_amdgcn_s_wait_tensorcnt(0);
    } else if (wv == 1) {
      tdm_load_tile(&lA[0][0], A + (size_t)tileM * lda + k0, 128, 64, lda);
      __builtin_amdgcn_s_wait_tensorcnt(0);
    }
    __syncthreads();                  // DMA'd tiles visible to all waves
    #pragma unroll
    for (int kk = 0; kk < 4; ++kk) {
      v16bf af;
      #pragma unroll
      for (int i = 0; i < 16; ++i)
        af[i] = lA[rblk * 16 + nlane][kk * 32 + khalf * 16 + i];
      #pragma unroll
      for (int j = 0; j < 4; ++j) {
        v16bf bv;
        #pragma unroll
        for (int i = 0; i < 16; ++i)
          bv[i] = lBt[cbase + j * 16 + nlane][kk * 32 + khalf * 16 + i];
        acc[j] = __builtin_amdgcn_wmma_f32_16x16x32_bf16(false, af, false, bv,
                                                         (short)0, acc[j], false, false);
      }
    }
  }

  #pragma unroll
  for (int j = 0; j < 4; ++j) {
    #pragma unroll
    for (int e = 0; e < 8; ++e) {
      int row = tileM + rblk * 16 + khalf * 8 + e;
      int col = tileN + cbase + j * 16 + nlane;
      size_t idx = (size_t)row * N + col;
      float vv = acc[j][e];
      if (bias) vv += bias[col];
      if      (EPI == 0) ((float*)Cout)[idx] = vv;
      else if (EPI == 1) ((__bf16*)Cout)[idx] = (__bf16)vv;
      else if (EPI == 2) ((float*)Cout)[idx] = sigf(vv);
      else if (EPI == 3) ((float*)Cout)[idx] = sigf(aux1[idx]) * vv;
      else if (EPI == 4) ((float*)Cout)[idx] = sigf(aux1[idx]) * vv + aux2[idx];
      else if (EPI == 5) { float hx = aux1[idx];
                           ((__bf16*)Cout)[idx] = (__bf16)(hx * sigf(hx) * vv); }
    }
  }
}

// ---------------------------------------------------------------------------
// Local attention, one (block b, head h) per 128-thread workgroup.
// scores[32,128] = (Q Kt)/sqrt(D) + zb + kbias ; softmax ; out = P V * gate
// ---------------------------------------------------------------------------
__global__ __launch_bounds__(128) void attn_kernel(
    const __bf16* __restrict__ qb, const __bf16* __restrict__ kb,
    const __bf16* __restrict__ vb, const float*  __restrict__ g,
    const __bf16* __restrict__ zb, const float*  __restrict__ kbias,
    __bf16* __restrict__ og)
{
  __shared__ __bf16 lq [32][40];
  __shared__ __bf16 lk [128][40];
  __shared__ __bf16 lvT[32][136];   // [dim][key]
  __shared__ float  lsc[32][132];
  __shared__ __bf16 lp [32][136];
  __shared__ float  red[32][4];

  const int b = blockIdx.x, h = blockIdx.y;
  const int tid = threadIdx.x;
  const int wv = tid >> 5, lane = tid & 31;
  const int nlane = lane & 15, khalf = lane >> 4;

  { // Q: 32 rows x 32 dims
    int r = tid >> 2, c8 = (tid & 3) * 8;
    uint4 d = *reinterpret_cast<const uint4*>(&qb[(size_t)(b * 32 + r) * 128 + h * 32 + c8]);
    *reinterpret_cast<uint4*>(&lq[r][c8]) = d;
  }
  #pragma unroll
  for (int i = 0; i < 4; ++i) { // K, V (V transposed): 128 window rows
    int c = tid + i * 128;
    int r = c >> 2, c8 = (c & 3) * 8;
    int at = b * 32 - 48 + r;
    at = at < 0 ? 0 : (at >= N_ATOM ? N_ATOM - 1 : at);
    uint4 dk = *reinterpret_cast<const uint4*>(&kb[(size_t)at * 128 + h * 32 + c8]);
    *reinterpret_cast<uint4*>(&lk[r][c8]) = dk;
    uint4 dv = *reinterpret_cast<const uint4*>(&vb[(size_t)at * 128 + h * 32 + c8]);
    const __bf16* pv = reinterpret_cast<const __bf16*>(&dv);
    #pragma unroll
    for (int jj = 0; jj < 8; ++jj) lvT[c8 + jj][r] = pv[jj];
  }
  __syncthreads();

  const float inv_sqrt_d = 0.17677669529663687f; // 1/sqrt(32)
  const __bf16* zrow = zb + (size_t)(b * HH + h) * (NQ * NK);

  #pragma unroll
  for (int it = 0; it < 4; ++it) {   // 16 score tiles, 4 per wave
    int t2 = wv * 4 + it;
    int qt = t2 >> 3, kt = t2 & 7;
    v16bf af, bv;
    #pragma unroll
    for (int i = 0; i < 16; ++i)
      af[i] = lq[qt * 16 + nlane][khalf * 16 + i];
    #pragma unroll
    for (int i = 0; i < 16; ++i)
      bv[i] = lk[kt * 16 + nlane][khalf * 16 + i];   // B = K^T : [dim][key]
    v8f acc = {};
    acc = __builtin_amdgcn_wmma_f32_16x16x32_bf16(false, af, false, bv, (short)0, acc, false, false);
    #pragma unroll
    for (int e = 0; e < 8; ++e) {
      int qr = qt * 16 + khalf * 8 + e;
      int kc = kt * 16 + nlane;
      lsc[qr][kc] = acc[e] * inv_sqrt_d + (float)zrow[qr * 128 + kc] + kbias[b * 128 + kc];
    }
  }
  __syncthreads();

  { // softmax over 128 keys, 4 threads per row
    int row = tid >> 2, part = tid & 3, c0 = part * 32;
    float m = -1e30f;
    for (int c = 0; c < 32; ++c) m = fmaxf(m, lsc[row][c0 + c]);
    red[row][part] = m; __syncthreads();
    float rm = fmaxf(fmaxf(red[row][0], red[row][1]), fmaxf(red[row][2], red[row][3]));
    __syncthreads();
    float s = 0.0f;
    for (int c = 0; c < 32; ++c) { float e = __expf(lsc[row][c0 + c] - rm); lsc[row][c0 + c] = e; s += e; }
    red[row][part] = s; __syncthreads();
    float inv = 1.0f / (red[row][0] + red[row][1] + red[row][2] + red[row][3]);
    for (int c = 0; c < 32; ++c) lp[row][c0 + c] = (__bf16)(lsc[row][c0 + c] * inv);
  }
  __syncthreads();

  { // out = P @ V, one 16x16 tile per wave, K=128 in 4 steps
    int qt = wv >> 1, dt = wv & 1;
    v8f acc = {};
    #pragma unroll
    for (int ks = 0; ks < 4; ++ks) {
      v16bf af, bv;
      #pragma unroll
      for (int i = 0; i < 16; ++i)
        af[i] = lp[qt * 16 + nlane][ks * 32 + khalf * 16 + i];
      #pragma unroll
      for (int i = 0; i < 16; ++i)
        bv[i] = lvT[dt * 16 + nlane][ks * 32 + khalf * 16 + i];
      acc = __builtin_amdgcn_wmma_f32_16x16x32_bf16(false, af, false, bv, (short)0, acc, false, false);
    }
    #pragma unroll
    for (int e = 0; e < 8; ++e) {
      int qr = qt * 16 + khalf * 8 + e;
      int dcol = dt * 16 + nlane;
      size_t idx = (size_t)(b * 32 + qr) * 128 + h * 32 + dcol;
      og[idx] = (__bf16)(acc[e] * g[idx]);
    }
  }
}

// ---------------------------------------------------------------------------
// Elementwise / row kernels
// ---------------------------------------------------------------------------
__global__ void k_cast_bf16(const float* __restrict__ in, __bf16* __restrict__ out, int n) {
  int i = blockIdx.x * 256 + threadIdx.x;
  if (i < n) out[i] = (__bf16)in[i];
}

// Weight prep: out[n*K+k] = bf16( W[k*N+n] * (scale ? scale[k] : 1) )
__global__ void k_wprep(const float* __restrict__ W, const float* __restrict__ scale,
                        __bf16* __restrict__ out, int K, int N) {
  int i = blockIdx.x * 256 + threadIdx.x;
  if (i >= K * N) return;
  int n = i / K, k = i - n * K;
  float s = scale ? scale[k] : 1.0f;
  out[i] = (__bf16)(W[(size_t)k * N + n] * s);
}

__global__ __launch_bounds__(128) void k_layernorm_row(
    const float* __restrict__ in, __bf16* __restrict__ ob) {
  __shared__ float sbuf[128];
  int row = blockIdx.x, t = threadIdx.x;
  size_t idx = (size_t)row * 128 + t;
  float x = in[idx];
  sbuf[t] = x; __syncthreads();
  for (int s = 64; s > 0; s >>= 1) { if (t < s) sbuf[t] += sbuf[t + s]; __syncthreads(); }
  float mu = sbuf[0] * (1.0f / 128.0f); __syncthreads();
  float d = x - mu;
  sbuf[t] = d * d; __syncthreads();
  for (int s = 64; s > 0; s >>= 1) { if (t < s) sbuf[t] += sbuf[t + s]; __syncthreads(); }
  ob[idx] = (__bf16)(d * rsqrtf(sbuf[0] * (1.0f / 128.0f) + 1e-5f));
}

__global__ void k_gather_add(const float* __restrict__ ql, const float* __restrict__ tok,
                             const int* __restrict__ a2t, const float* __restrict__ tmask,
                             const float* __restrict__ amask, float* __restrict__ a) {
  int i = blockIdx.x * 256 + threadIdx.x;
  int atom = i >> 7, c = i & 127;
  int t = a2t[atom];
  a[i] = ql[i] + tok[t * 128 + c] * tmask[t] * amask[atom];
}

__global__ void k_kbias(const float* __restrict__ amask, float* __restrict__ kbias) {
  int i = blockIdx.x * 256 + threadIdx.x;       // NB*NK
  int b = i >> 7, j = i & 127;
  int kidx = b * 32 - 48 + j;
  bool in = (kidx >= 0) && (kidx < N_ATOM);
  int kc = kidx < 0 ? 0 : (kidx >= N_ATOM ? N_ATOM - 1 : kidx);
  float m = in ? amask[kc] : 0.0f;
  kbias[i] = (m - 1.0f) * 1e9f;
}

// zb[l,b,h,q,k] for ALL layers from one pass over plm (ln over C_PAIR is layer-
// invariant; per-layer affine+w_pair folded to a 16x(L*H) matrix).
__global__ __launch_bounds__(256) void k_zbias(
    const float* __restrict__ plm, const float* __restrict__ ln_z_w,
    const float* __restrict__ ln_z_b, const float* __restrict__ w_pair,
    __bf16* __restrict__ zb) {
  __shared__ float fw[3][16][4];
  __shared__ float fb[3][4];
  int t = threadIdx.x;
  if (t < 192) {
    int l = t >> 6, rem = t & 63, c = rem >> 2, h = rem & 3;
    fw[l][c][h] = ln_z_w[l * 16 + c] * w_pair[(l * 16 + c) * 4 + h];
  } else if (t < 204) {
    int u = t - 192, l = u >> 2, h = u & 3;
    float s = 0.0f;
    for (int c = 0; c < 16; ++c) s += ln_z_b[l * 16 + c] * w_pair[(l * 16 + c) * 4 + h];
    fb[l][h] = s;
  }
  __syncthreads();
  size_t gid = (size_t)blockIdx.x * 256 + t;    // < NB*NQ*NK
  float p[16];
  float mu = 0.0f;
  #pragma unroll
  for (int c = 0; c < 16; ++c) { p[c] = plm[gid * 16 + c]; mu += p[c]; }
  mu *= (1.0f / 16.0f);
  float var = 0.0f;
  #pragma unroll
  for (int c = 0; c < 16; ++c) { p[c] -= mu; var += p[c] * p[c]; }
  float inv = rsqrtf(var * (1.0f / 16.0f) + 1e-5f);
  int b  = (int)(gid >> 12);
  int qk = (int)(gid & 4095);
  #pragma unroll
  for (int l = 0; l < 3; ++l)
    #pragma unroll
    for (int h = 0; h < 4; ++h) {
      float s = fb[l][h];
      #pragma unroll
      for (int c = 0; c < 16; ++c) s += p[c] * inv * fw[l][c][h];
      zb[(((size_t)l * NB + b) * HH + h) * 4096 + qk] = (__bf16)s;
    }
}

// x = sigmoid(gate_pre) * ln(a) + skip   (bf16 out)
__global__ __launch_bounds__(128) void k_adaln_combine(
    const float* __restrict__ a, const float* __restrict__ gate_pre,
    const float* __restrict__ skip, __bf16* __restrict__ x) {
  __shared__ float sbuf[128];
  int row = blockIdx.x, t = threadIdx.x;
  size_t idx = (size_t)row * 128 + t;
  float v = a[idx];
  sbuf[t] = v; __syncthreads();
  for (int s = 64; s > 0; s >>= 1) { if (t < s) sbuf[t] += sbuf[t + s]; __syncthreads(); }
  float mu = sbuf[0] * (1.0f / 128.0f); __syncthreads();
  float d = v - mu;
  sbuf[t] = d * d; __syncthreads();
  for (int s = 64; s > 0; s >>= 1) { if (t < s) sbuf[t] += sbuf[t + s]; __syncthreads(); }
  float nrm = d * rsqrtf(sbuf[0] * (1.0f / 128.0f) + 1e-5f);
  x[idx] = (__bf16)(sigf(gate_pre[idx]) * nrm + skip[idx]);
}

__global__ __launch_bounds__(128) void k_final(
    const float* __restrict__ a, const float* __restrict__ ln_w, const float* __restrict__ ln_b,
    const float* __restrict__ w_out, float* __restrict__ out) {
  __shared__ float sbuf[128];
  int row = blockIdx.x, t = threadIdx.x;
  size_t idx = (size_t)row * 128 + t;
  float v = a[idx];
  sbuf[t] = v; __syncthreads();
  for (int s = 64; s > 0; s >>= 1) { if (t < s) sbuf[t] += sbuf[t + s]; __syncthreads(); }
  float mu = sbuf[0] * (1.0f / 128.0f); __syncthreads();
  float d = v - mu;
  sbuf[t] = d * d; __syncthreads();
  for (int s = 64; s > 0; s >>= 1) { if (t < s) sbuf[t] += sbuf[t + s]; __syncthreads(); }
  float y = d * rsqrtf(sbuf[0] * (1.0f / 128.0f) + 1e-5f) * ln_w[t] + ln_b[t];
  __syncthreads();
  for (int o = 0; o < 3; ++o) {
    sbuf[t] = y * w_out[t * 3 + o]; __syncthreads();
    for (int s = 64; s > 0; s >>= 1) { if (t < s) sbuf[t] += sbuf[t + s]; __syncthreads(); }
    if (t == 0) out[row * 3 + o] = sbuf[0];
    __syncthreads();
  }
}

// ---------------------------------------------------------------------------
extern "C" void kernel_launch(void* const* d_in, const int* in_sizes, int n_in,
                              void* d_out, int out_size, void* d_ws, size_t ws_size,
                              hipStream_t stream) {
  (void)in_sizes; (void)n_in; (void)out_size; (void)ws_size;
  const float* ai          = (const float*)d_in[0];
  const float* ql          = (const float*)d_in[1];
  const float* cl          = (const float*)d_in[2];
  const float* plm         = (const float*)d_in[3];
  const float* token_mask  = (const float*)d_in[4];
  const float* atom_mask   = (const float*)d_in[5];
  const int*   a2t         = (const int*)d_in[6];
  const float* w_q_in      = (const float*)d_in[8];
  const float* attn_ln_s_w = (const float*)d_in[9];
  const float* attn_gate_w = (const float*)d_in[10];
  const float* attn_gate_b = (const float*)d_in[11];
  const float* attn_skip_w = (const float*)d_in[12];
  const float* wq          = (const float*)d_in[13];
  const float* bq          = (const float*)d_in[14];
  const float* wk          = (const float*)d_in[15];
  const float* wvv         = (const float*)d_in[16];
  const float* ln_z_w      = (const float*)d_in[17];
  const float* ln_z_b      = (const float*)d_in[18];
  const float* w_pair      = (const float*)d_in[19];
  const float* w_gate      = (const float*)d_in[20];
  const float* w_o         = (const float*)d_in[21];
  const float* w_sg        = (const float*)d_in[22];
  const float* b_sg        = (const float*)d_in[23];
  const float* tr_ln_s_w   = (const float*)d_in[24];
  const float* tr_gate_w   = (const float*)d_in[25];
  const float* tr_gate_b   = (const float*)d_in[26];
  const float* tr_skip_w   = (const float*)d_in[27];
  const float* w_swish     = (const float*)d_in[28];
  const float* w_lin       = (const float*)d_in[29];
  const float* w_down      = (const float*)d_in[30];
  const float* tr_sg_w     = (const float*)d_in[31];
  const float* tr_sg_b     = (const float*)d_in[32];
  const float* ln_w        = (const float*)d_in[33];
  const float* ln_b        = (const float*)d_in[34];
  const float* w_out       = (const float*)d_in[35];

  // workspace bump allocator (~140 MB total)
  char* ws = (char*)d_ws;
  size_t off = 0;
  auto alloc = [&](size_t bytes) { void* p = ws + off; off = (off + bytes + 255) & ~(size_t)255; return p; };
  __bf16* ai_bf = (__bf16*)alloc((size_t)N_TOKEN * C_TOKEN * 2);
  float*  tok   = (float*) alloc((size_t)N_TOKEN * C_ATOM * 4);
  __bf16* sn_b  = (__bf16*)alloc((size_t)N_ATOM * 128 * 2);
  float*  abuf  = (float*) alloc((size_t)N_ATOM * 128 * 4);
  __bf16* xb    = (__bf16*)alloc((size_t)N_ATOM * 128 * 2);
  __bf16* qb    = (__bf16*)alloc((size_t)N_ATOM * 128 * 2);
  __bf16* kbuf  = (__bf16*)alloc((size_t)N_ATOM * 128 * 2);
  __bf16* vbuf  = (__bf16*)alloc((size_t)N_ATOM * 128 * 2);
  float*  gbuf  = (float*) alloc((size_t)N_ATOM * 128 * 4);
  __bf16* ogb   = (__bf16*)alloc((size_t)N_ATOM * 128 * 2);
  float*  batt  = (float*) alloc((size_t)N_ATOM * 128 * 4);
  float*  t1    = (float*) alloc((size_t)N_ATOM * 128 * 4);
  float*  t2    = (float*) alloc((size_t)N_ATOM * 128 * 4);
  float*  t256  = (float*) alloc((size_t)N_ATOM * 256 * 4);
  __bf16* hid   = (__bf16*)alloc((size_t)N_ATOM * 256 * 2);
  float*  kbias = (float*) alloc((size_t)NB * NK * 4);
  __bf16* zb    = (__bf16*)alloc((size_t)LL * NB * HH * NQ * NK * 2);

  // ---- weight prep: f32 [K,N] -> bf16 [N,K], AdaLN row-scales folded in ----
  auto prep = [&](const float* W, const float* scale, int K, int N) -> __bf16* {
    __bf16* dst = (__bf16*)alloc((size_t)K * N * 2);
    k_wprep<<<(K * N + 255) / 256, 256, 0, stream>>>(W, scale, dst, K, N);
    return dst;
  };
  __bf16* Wqin = prep(w_q_in, nullptr, C_TOKEN, 128);
  __bf16 *Wga[LL], *Wsk[LL], *Wq[LL], *Wk[LL], *Wv[LL], *Wg[LL], *Wsg[LL], *Wo[LL];
  __bf16 *Wtg[LL], *Wts[LL], *Wtsg[LL], *Wsw[LL], *Wli[LL], *Wdn[LL];
  for (int l = 0; l < LL; ++l) {
    const size_t o128 = (size_t)l * 128 * 128, o256 = (size_t)l * 128 * 256, ob = (size_t)l * 128;
    Wga[l]  = prep(attn_gate_w + o128, attn_ln_s_w + ob, 128, 128);
    Wsk[l]  = prep(attn_skip_w + o128, attn_ln_s_w + ob, 128, 128);
    Wq[l]   = prep(wq + o128,    nullptr, 128, 128);
    Wk[l]   = prep(wk + o128,    nullptr, 128, 128);
    Wv[l]   = prep(wvv + o128,   nullptr, 128, 128);
    Wg[l]   = prep(w_gate + o128, nullptr, 128, 128);
    Wsg[l]  = prep(w_sg + o128,  nullptr, 128, 128);
    Wo[l]   = prep(w_o + o128,   nullptr, 128, 128);
    Wtg[l]  = prep(tr_gate_w + o128, tr_ln_s_w + ob, 128, 128);
    Wts[l]  = prep(tr_skip_w + o128, tr_ln_s_w + ob, 128, 128);
    Wtsg[l] = prep(tr_sg_w + o128, nullptr, 128, 128);
    Wsw[l]  = prep(w_swish + o256, nullptr, 128, 256);
    Wli[l]  = prep(w_lin + o256,   nullptr, 128, 256);
    Wdn[l]  = prep(w_down + o256,  nullptr, 256, 128);
  }

  // ---- preprocessing ----
  k_cast_bf16<<<(N_TOKEN * C_TOKEN) / 256, 256, 0, stream>>>(ai, ai_bf, N_TOKEN * C_TOKEN);
  gemm_bf16<0><<<dim3(N_TOKEN / 64, 1), 256, 0, stream>>>(
      ai_bf, C_TOKEN, Wqin, C_TOKEN, nullptr, nullptr, nullptr,
      tok, N_TOKEN, C_ATOM, C_TOKEN);
  k_layernorm_row<<<N_ATOM, 128, 0, stream>>>(cl, sn_b);
  k_gather_add<<<(N_ATOM * 128) / 256, 256, 0, stream>>>(ql, tok, a2t, token_mask, atom_mask, abuf);
  k_kbias<<<(NB * NK) / 256, 256, 0, stream>>>(atom_mask, kbias);
  k_zbias<<<(NB * NQ * NK) / 256, 256, 0, stream>>>(plm, ln_z_w, ln_z_b, w_pair, zb);

  dim3 g64(N_ATOM / 64, 1);
  dim3 g64x2(N_ATOM / 64, 2);
  const size_t zlayer = (size_t)NB * HH * NQ * NK;

  for (int l = 0; l < LL; ++l) {
    const size_t ob = (size_t)l * 128;
    // AdaLN for attention: gate_pre, skip (ln_s_w pre-folded into weights)
    gemm_bf16<0><<<g64, 256, 0, stream>>>(sn_b, 128, Wga[l], 128,
        attn_gate_b + ob, nullptr, nullptr, t1, N_ATOM, 128, 128);
    gemm_bf16<0><<<g64, 256, 0, stream>>>(sn_b, 128, Wsk[l], 128,
        nullptr, nullptr, nullptr, t2, N_ATOM, 128, 128);
    k_adaln_combine<<<N_ATOM, 128, 0, stream>>>(abuf, t1, t2, xb);
    // projections
    gemm_bf16<1><<<g64, 256, 0, stream>>>(xb, 128, Wq[l], 128,
        bq + ob, nullptr, nullptr, qb, N_ATOM, 128, 128);
    gemm_bf16<1><<<g64, 256, 0, stream>>>(xb, 128, Wk[l], 128,
        nullptr, nullptr, nullptr, kbuf, N_ATOM, 128, 128);
    gemm_bf16<1><<<g64, 256, 0, stream>>>(xb, 128, Wv[l], 128,
        nullptr, nullptr, nullptr, vbuf, N_ATOM, 128, 128);
    gemm_bf16<2><<<g64, 256, 0, stream>>>(xb, 128, Wg[l], 128,
        nullptr, nullptr, nullptr, gbuf, N_ATOM, 128, 128);
    gemm_bf16<0><<<g64, 256, 0, stream>>>(sn_b, 128, Wsg[l], 128,
        b_sg + ob, nullptr, nullptr, t1, N_ATOM, 128, 128);
    // local attention (gate fused into store)
    attn_kernel<<<dim3(NB, HH), 128, 0, stream>>>(qb, kbuf, vbuf, gbuf,
        zb + (size_t)l * zlayer, kbias, ogb);
    // b_att = sigmoid(sg_pre) * ((o*g) @ w_o)
    gemm_bf16<3><<<g64, 256, 0, stream>>>(ogb, 128, Wo[l], 128,
        nullptr, t1, nullptr, batt, N_ATOM, 128, 128);
    // transition
    gemm_bf16<0><<<g64, 256, 0, stream>>>(sn_b, 128, Wtg[l], 128,
        tr_gate_b + ob, nullptr, nullptr, t1, N_ATOM, 128, 128);
    gemm_bf16<0><<<g64, 256, 0, stream>>>(sn_b, 128, Wts[l], 128,
        nullptr, nullptr, nullptr, t2, N_ATOM, 128, 128);
    k_adaln_combine<<<N_ATOM, 128, 0, stream>>>(abuf, t1, t2, xb);
    gemm_bf16<0><<<g64x2, 256, 0, stream>>>(xb, 128, Wsw[l], 128,
        nullptr, nullptr, nullptr, t256, N_ATOM, 256, 128);
    gemm_bf16<5><<<g64x2, 256, 0, stream>>>(xb, 128, Wli[l], 128,
        nullptr, t256, nullptr, hid, N_ATOM, 256, 128);
    gemm_bf16<0><<<g64, 256, 0, stream>>>(sn_b, 128, Wtsg[l], 128,
        tr_sg_b + ob, nullptr, nullptr, t1, N_ATOM, 128, 128);
    // a = sigmoid(tsg_pre) * (hidden @ w_down) + b_att
    gemm_bf16<4><<<g64, 256, 0, stream>>>(hid, 256, Wdn[l], 256,
        nullptr, t1, batt, abuf, N_ATOM, 128, 256);
  }

  k_final<<<N_ATOM, 128, 0, stream>>>(abuf, ln_w, ln_b, w_out, (float*)d_out);
}
